// Diff_52338471469225
// MI455X (gfx1250) — compile-verified, matrix operand order checked by reading, not run on hardware
//
#include <hip/hip_runtime.h>

// ---------------------------------------------------------------------------
// CDNA5 (gfx1250, wave32) fused self-attention:
//   q = Wq*F3+bq (32x256), k = Wk*F1+bk (32x256), v = Wv*F2+bv (256x256)
//   out = gamma * ( V * softmax(q^T k / sqrt(32))^T ) + F3
// B=4, C=256, N=4096, Cq=32.
// bf16 WMMA (v_wmma_f32_16x16x32_bf16) for all GEMMs; flash-style fusion keeps
// the 268MB score matrix out of global memory (only 64x64 tiles in LDS); K/V tiles streamed from L2 with CDNA5
// async-to-LDS DMA, double-buffered so DMA overlaps the compute phase.
// ---------------------------------------------------------------------------

typedef __attribute__((ext_vector_type(16))) __bf16 v16bf;
typedef __attribute__((ext_vector_type(8)))  float  v8f;

#define WAVE_LANE (threadIdx.x & 31)

// CDNA5 async global->LDS copy (GLOBAL_LOAD_ASYNC_TO_LDS_B128, ASYNCcnt).
__device__ inline void async_load_b128(const void* lds_ptr, const void* gaddr) {
  unsigned lds_off = (unsigned)(uintptr_t)lds_ptr;
  asm volatile("global_load_async_to_lds_b128 %0, %1, off"
               :: "v"(lds_off), "v"((unsigned long long)(uintptr_t)gaddr)
               : "memory");
}
__device__ inline void wait_async0() {
  asm volatile("s_wait_asynccnt 0x0" ::: "memory");
}

// --- A-fragment (16x32 bf16, M rows x K cols), rows contiguous in LDS with
// leading dim `ld` (elements, multiple of 8 => 16B-aligned vector loads).
// ISA 7.12.2: lanes 0-15 hold M=lane, elems 0..7=K0..7, 8..15=K16..23;
// lanes 16-31 hold M=lane-16, elems 0..7=K8..15, 8..15=K24..31.
__device__ inline v16bf frag_A(const __bf16* base, int ld) {
  const int lane = WAVE_LANE;
  const int m    = lane & 15;
  const int koff = (lane >> 4) << 3;  // 0 or 8
  const __bf16* p = base + m * ld + koff;
  union { uint4 u[2]; v16bf v; } r;
  r.u[0] = *reinterpret_cast<const uint4*>(p);
  r.u[1] = *reinterpret_cast<const uint4*>(p + 16);
  return r.v;
}

// --- B-fragment (32x16 bf16, K rows x N cols) from a [K][N] LDS tile.
__device__ inline v16bf frag_B(const __bf16* base, int ld) {
  const int lane = WAVE_LANE;
  const int n    = lane & 15;
  const int koff = (lane >> 4) << 3;
  v16bf b;
#pragma unroll
  for (int e = 0; e < 8; ++e) b[e]     = base[(koff + e) * ld + n];
#pragma unroll
  for (int e = 0; e < 8; ++e) b[e + 8] = base[(koff + 16 + e) * ld + n];
  return b;
}

// ===========================================================================
// Kernel 1: projection GEMM.  Y[b,o,n] = sum_c W[o,c]*X[b,c,n] + bias[o]
// Block = 256 threads (8 waves) -> [O x 64] bf16 output tile for batch b.
// ===========================================================================
template <int O>
__global__ __launch_bounds__(256) void proj_kernel(const float* __restrict__ X,
                                                   const float* __restrict__ W,
                                                   const float* __restrict__ bias,
                                                   __bf16* __restrict__ Y) {
  constexpr int LDW = 40;
  constexpr int LDX = 72;
  constexpr int TPW = O / 32;  // D-tiles per wave: O=256 -> 8, O=32 -> 1

  __shared__ __bf16 sW[O * LDW];   // [o][k 0..32) chunk
  __shared__ __bf16 sX[32 * LDX];  // [k][n 0..64) chunk

  const int b   = blockIdx.y;
  const int n0  = blockIdx.x * 64;
  const int tid = threadIdx.x;
  const int lane = tid & 31, w = tid >> 5;

  v8f acc[TPW];
#pragma unroll
  for (int i = 0; i < TPW; ++i)
#pragma unroll
    for (int r = 0; r < 8; ++r) acc[i][r] = 0.0f;

  const float* Xb = X + (size_t)b * 256 * 4096 + n0;

  for (int kc = 0; kc < 256; kc += 32) {
    __syncthreads();
    for (int idx = tid; idx < O * 32; idx += 256) {  // W chunk f32 -> bf16
      int o = idx >> 5, k = idx & 31;
      sW[o * LDW + k] = (__bf16)W[(size_t)o * 256 + kc + k];
    }
    {  // X chunk: row k = tid>>3, 8-elem run (coalesced 256B per 8 lanes)
      int k = tid >> 3, nch = (tid & 7) * 8;
#pragma unroll
      for (int e = 0; e < 8; ++e)
        sX[k * LDX + nch + e] = (__bf16)Xb[(size_t)(kc + k) * 4096 + nch + e];
    }
    __syncthreads();

#pragma unroll
    for (int i = 0; i < TPW; ++i) {
      const int t = w * TPW + i;
      const int ot = t >> 2, nt = t & 3;
      v16bf A = frag_A(&sW[ot * 16 * LDW], LDW);
      v16bf B = frag_B(&sX[nt * 16], LDX);
      acc[i] = __builtin_amdgcn_wmma_f32_16x16x32_bf16(
          false, A, false, B, (short)0, acc[i], false, false);
    }
  }

  __bf16* Yb = Y + (size_t)b * O * 4096 + n0;
#pragma unroll
  for (int i = 0; i < TPW; ++i) {
    const int t = w * TPW + i;
    const int ot = t >> 2, nt = t & 3;
    const int n = nt * 16 + (lane & 15);
#pragma unroll
    for (int r = 0; r < 8; ++r) {
      const int o = ot * 16 + ((lane >> 4) << 3) + r;
      Yb[(size_t)o * 4096 + n] = (__bf16)(acc[i][r] + bias[o]);
    }
  }
}

// ===========================================================================
// Kernel 2: fused flash attention + residual epilogue.
// Block = 256 threads (8 waves) owns (batch b, 64 query columns).
// Double-buffered 64-wide K/V tiles via async-to-LDS DMA (DMA of tile t+1
// overlaps compute of tile t); online softmax in LDS.
// Wave w accumulates output rows c in [32w, 32w+32): acc[2][4] 16x16 tiles.
// ===========================================================================
__global__ __launch_bounds__(256) void attn_kernel(const __bf16* __restrict__ qg,
                                                   const __bf16* __restrict__ kg,
                                                   const __bf16* __restrict__ vg,
                                                   const float* __restrict__ F3,
                                                   const float* __restrict__ gamma_p,
                                                   float* __restrict__ out) {
  constexpr int NT = 64, MT = 64, NTILES = 4096 / MT;
  constexpr int LDQ = 40, LDK = 72, LDV = 72, LDS_S = 68, LDP = 72;
  constexpr int KBUF = 32 * LDK;    // 4608B per K buffer (bf16 elems)
  constexpr int VBUF = 256 * LDV;   // 36864B per V buffer

  extern __shared__ char smem[];
  __bf16* sQ  = (__bf16*)(smem);              // [64][40] q^T (n,c)    5120B
  __bf16* sKb = (__bf16*)(smem + 5120);       // 2x [32][72] k (c,m)   9216B
  __bf16* sVb = (__bf16*)(smem + 14336);      // 2x [256][72] v (c,m) 73728B
  float*  sS  = (float*)(smem + 88064);       // [64][68] scores      17408B
  __bf16* sP  = (__bf16*)(smem + 105472);     // [64][72] probs bf16   9216B
  float*  sMax   = (float*)(smem + 114688);   // [64]
  float*  sSum   = (float*)(smem + 114944);   // [64]
  float*  sAlpha = (float*)(smem + 115200);   // [64]
  float*  sRed   = (float*)(smem + 115456);   // [4][64] partials      1024B
                                              // total 116480B (2 blk/WGP)
  const int b   = blockIdx.y;
  const int n0  = blockIdx.x * NT;
  const int tid = threadIdx.x;
  const int lane = tid & 31, w = tid >> 5;

  const __bf16* qb = qg + (size_t)b * 32 * 4096 + n0;
  const __bf16* kb = kg + (size_t)b * 32 * 4096;
  const __bf16* vb = vg + (size_t)b * 256 * 4096;
  const float scale = 0.17677669529663687f;  // 1/sqrt(Cq=32)

  // Issue async DMA for K/V tile starting at column m0 into buffer p.
  auto stage = [&](int m0, int p) {
    __bf16* dK = sKb + p * KBUF;
    __bf16* dV = sVb + p * VBUF;
    {
      int c = tid >> 3, off = (tid & 7) * 8;
      async_load_b128(&dK[c * LDK + off], kb + (size_t)c * 4096 + m0 + off);
    }
#pragma unroll
    for (int it = 0; it < 8; ++it) {
      int ch = tid + it * 256;
      int r = ch >> 3, off = (ch & 7) * 8;
      const __bf16* src = vb + (size_t)r * 4096 + m0 + off;
      async_load_b128(&dV[r * LDV + off], src);
      if (m0 + MT < 4096) __builtin_prefetch(src + MT, 0, 1);  // warm L2
    }
  };

  // Stage q transposed: sQ[n][c] so QK A-fragments are contiguous-K loads.
  {
    int c = tid >> 3, nch = (tid & 7) * 8;
#pragma unroll
    for (int e = 0; e < 8; ++e)
      sQ[(nch + e) * LDQ + c] = qb[(size_t)c * 4096 + nch + e];
  }
  if (tid < NT) { sMax[tid] = -1e30f; sSum[tid] = 0.0f; }

  v8f acc[2][4];
#pragma unroll
  for (int i = 0; i < 2; ++i)
#pragma unroll
    for (int j = 0; j < 4; ++j)
#pragma unroll
      for (int r = 0; r < 8; ++r) acc[i][j][r] = 0.0f;

  stage(0, 0);  // prologue: DMA first tile

  for (int t = 0; t < NTILES; ++t) {
    const int p = t & 1;
    const __bf16* cK = sKb + p * KBUF;
    const __bf16* cV = sVb + p * VBUF;

    wait_async0();    // tile t landed in buffer p
    __syncthreads();  // publish; also: everyone done reading buffer p^1
    if (t + 1 < NTILES) stage((t + 1) * MT, p ^ 1);  // overlap DMA w/ compute

    // ---- S = q^T k: 16 tiles of 16x16 (K=32 => one WMMA each), 2/wave ----
#pragma unroll
    for (int i = 0; i < 2; ++i) {
      const int tt = w * 2 + i, sn = tt >> 2, sm = tt & 3;
      v16bf A = frag_A(&sQ[sn * 16 * LDQ], LDQ);
      v16bf B = frag_B(&cK[sm * 16], LDK);
      v8f d;
#pragma unroll
      for (int r = 0; r < 8; ++r) d[r] = 0.0f;
      d = __builtin_amdgcn_wmma_f32_16x16x32_bf16(
          false, A, false, B, (short)0, d, false, false);
#pragma unroll
      for (int r = 0; r < 8; ++r)  // D: col = lane&15, row = 8*(lane>>4)+r
        sS[(sn * 16 + ((lane >> 4) << 3) + r) * LDS_S + sm * 16 + (lane & 15)] =
            d[r] * scale;
    }
    __syncthreads();

    // ---- online softmax, 4-way parallel per row (all 8 waves active) ----
    {  // phase A: partial max over 16-col slice
      const int row = tid & 63, qd = tid >> 6;
      const float* srow = &sS[row * LDS_S + qd * 16];
      float pmax = -1e30f;
#pragma unroll
      for (int m = 0; m < 16; ++m) pmax = fmaxf(pmax, srow[m]);
      sRed[qd * 64 + row] = pmax;
    }
    __syncthreads();
    if (tid < NT) {  // phase B: combine, running max + rescale factor
      const float mOld = sMax[tid];
      float mNew = fmaxf(fmaxf(sRed[tid], sRed[64 + tid]),
                         fmaxf(sRed[128 + tid], sRed[192 + tid]));
      mNew = fmaxf(mNew, mOld);
      sMax[tid]   = mNew;
      sAlpha[tid] = __expf(mOld - mNew);
    }
    __syncthreads();
    {  // phase C: exp + bf16 probs + partial sums
      const int row = tid & 63, qd = tid >> 6;
      const float mNew = sMax[row];
      const float* srow = &sS[row * LDS_S + qd * 16];
      __bf16* prow = &sP[row * LDP + qd * 16];
      float psum = 0.0f;
#pragma unroll
      for (int m = 0; m < 16; ++m) {
        float pv = __expf(srow[m] - mNew);
        psum += pv;
        prow[m] = (__bf16)pv;
      }
      sRed[qd * 64 + row] = psum;
    }
    __syncthreads();
    if (tid < NT)  // phase D: fold partial sums into running row sum
      sSum[tid] = sSum[tid] * sAlpha[tid] +
                  (sRed[tid] + sRed[64 + tid]) + (sRed[128 + tid] + sRed[192 + tid]);

    // ---- rescale accumulators (each v8f shares one column n) ----
#pragma unroll
    for (int j = 0; j < 4; ++j) {
      const float a = sAlpha[j * 16 + (lane & 15)];
#pragma unroll
      for (int i = 0; i < 2; ++i)
#pragma unroll
        for (int r = 0; r < 8; ++r) acc[i][j][r] *= a;
    }

    // ---- O[c,n] += sum_m V[c,m]*P[n,m]: A = V (K=m), B = P^T via frag_A ----
#pragma unroll
    for (int ks = 0; ks < MT; ks += 32) {
      v16bf Bf[4];
#pragma unroll
      for (int j = 0; j < 4; ++j) Bf[j] = frag_A(&sP[j * 16 * LDP + ks], LDP);
#pragma unroll
      for (int i = 0; i < 2; ++i) {
        const int crow = w * 32 + i * 16;
        v16bf A = frag_A(&cV[crow * LDV + ks], LDV);
#pragma unroll
        for (int j = 0; j < 4; ++j)
          acc[i][j] = __builtin_amdgcn_wmma_f32_16x16x32_bf16(
              false, A, false, Bf[j], (short)0, acc[i][j], false, false);
      }
    }
  }
  __syncthreads();  // publish final sSum (phase D) before epilogue reads

  // ---- epilogue: out = gamma * O/rowsum + F3 ----
  const float g = gamma_p[0];
  float* ob = out + (size_t)b * 256 * 4096 + n0;
  const float* f3 = F3 + (size_t)b * 256 * 4096 + n0;
#pragma unroll
  for (int j = 0; j < 4; ++j) {
    const int n = j * 16 + (lane & 15);
    const float inv = g / sSum[n];
#pragma unroll
    for (int i = 0; i < 2; ++i) {
#pragma unroll
      for (int r = 0; r < 8; ++r) {
        const int c = w * 32 + i * 16 + ((lane >> 4) << 3) + r;
        const size_t idx = (size_t)c * 4096 + n;
        ob[idx] = acc[i][j][r] * inv + f3[idx];
      }
    }
  }
}

// ===========================================================================
extern "C" void kernel_launch(void* const* d_in, const int* in_sizes, int n_in,
                              void* d_out, int out_size, void* d_ws, size_t ws_size,
                              hipStream_t stream) {
  const float* F3 = (const float*)d_in[0];
  const float* F1 = (const float*)d_in[1];
  const float* F2 = (const float*)d_in[2];
  const float* Wq = (const float*)d_in[3];
  const float* bq = (const float*)d_in[4];
  const float* Wk = (const float*)d_in[5];
  const float* bk = (const float*)d_in[6];
  const float* Wv = (const float*)d_in[7];
  const float* bv = (const float*)d_in[8];
  const float* gamma = (const float*)d_in[9];
  float* out = (float*)d_out;

  // Workspace: bf16 q [4][32][4096], k [4][32][4096], v [4][256][4096] ~10MB
  __bf16* qb = (__bf16*)d_ws;
  __bf16* kb = qb + (size_t)4 * 32 * 4096;
  __bf16* vb = kb + (size_t)4 * 32 * 4096;

  dim3 grid(64, 4);  // 64 n-tiles x 4 batches
  proj_kernel<32><<<grid, 256, 0, stream>>>(F3, Wq, bq, qb);
  proj_kernel<32><<<grid, 256, 0, stream>>>(F1, Wk, bk, kb);
  proj_kernel<256><<<grid, 256, 0, stream>>>(F2, Wv, bv, vb);
  attn_kernel<<<grid, 256, 116480, stream>>>(qb, kb, vb, F3, gamma, out);
}
